// MutiHeadSelection_29661044146243
// MI455X (gfx1250) — compile-verified, matrix operand order checked by reading, not run on hardware
//
#include <hip/hip_runtime.h>
#include <math.h>

// MI455X / gfx1250: wave32, WMMA 16x16x32 bf16 path.
typedef __attribute__((ext_vector_type(16))) __bf16 v16bf;
typedef __attribute__((ext_vector_type(2)))  __bf16 v2bf;
typedef __attribute__((ext_vector_type(8)))  float  v8f;
typedef unsigned int u32x4 __attribute__((ext_vector_type(4)));
typedef int          i32x8 __attribute__((ext_vector_type(8)));
typedef int          i32x4 __attribute__((ext_vector_type(4)));

#define NEGV 1000000000000.0f

static __device__ __forceinline__ unsigned short f2bfbits(float f) {
  union { float f; unsigned u; } x; x.f = f;
  unsigned u = x.u;
  unsigned r = u + 0x7FFFu + ((u >> 16) & 1u);   // round-to-nearest-even
  return (unsigned short)(r >> 16);
}

// ---------- packed fp32x2 -> bf16x2 ----------
// Preference: v_cvt_pk_bf16_f32 > v_perm_b32 pack (3 VALU/pair) > scalar RNE fallback.
static __device__ __forceinline__ v2bf pk2(float a, float b) {
#if __has_builtin(__builtin_amdgcn_cvt_pk_bf16_f32)
  return __builtin_amdgcn_cvt_pk_bf16_f32(a, b);
#else
  union { unsigned u; v2bf v; } t;
#if __has_builtin(__builtin_amdgcn_perm)
  // round-half-up then take high 16 bits of each float; pack with one v_perm_b32
  t.u = __builtin_amdgcn_perm(__float_as_uint(b) + 0x8000u,
                              __float_as_uint(a) + 0x8000u, 0x07060302u);
#else
  t.u = (unsigned)f2bfbits(a) | ((unsigned)f2bfbits(b) << 16);
#endif
  return t.v;
#endif
}

// ---------- tanh: hardware v_tanh_f32 > exp2/rcp TRANS pair > ocml ----------
static __device__ __forceinline__ float fast_tanh(float x) {
#if __has_builtin(__builtin_amdgcn_tanhf)
  return __builtin_amdgcn_tanhf(x);
#elif __has_builtin(__builtin_amdgcn_tanh_f32)
  return __builtin_amdgcn_tanh_f32(x);
#else
  const float ax = __builtin_fabsf(x);
  float e;
#if __has_builtin(__builtin_amdgcn_exp2f)
  e = __builtin_amdgcn_exp2f(-2.8853900817779268f * ax);  // exp(-2|x|)
#else
  e = exp2f(-2.8853900817779268f * ax);
#endif
  float r;
#if __has_builtin(__builtin_amdgcn_rcpf)
  r = (1.0f - e) * __builtin_amdgcn_rcpf(1.0f + e);
#else
  r = (1.0f - e) / (1.0f + e);
#endif
  return __builtin_copysignf(r, x);
#endif
}

// ---- prep: bf16 copies of Wh_w and zero-padded Wo_w (49x128 -> 64x128) ----
__global__ void prep_weights_kernel(const float* __restrict__ Whw,
                                    const float* __restrict__ Wow,
                                    unsigned short* __restrict__ whw_bf,
                                    unsigned short* __restrict__ wo_bf) {
  for (int i = threadIdx.x; i < 128 * 256; i += blockDim.x)
    whw_bf[i] = f2bfbits(Whw[i]);
  for (int i = threadIdx.x; i < 64 * 128; i += blockDim.x) {
    int c = i >> 7, k = i & 127;
    float v = (c < 49) ? Wow[c * 128 + k] : 0.0f;
    wo_bf[i] = f2bfbits(v);
  }
}

// ---- prep: x = inputs + sinusoid; proj_i = x@Whw[:,:128]^T + b; proj_j = x@Whw[:,128:]^T ----
__global__ void prep_proj_kernel(const float* __restrict__ inputs,
                                 const float* __restrict__ Whw,
                                 const float* __restrict__ Whb,
                                 float* __restrict__ proj_i,
                                 float* __restrict__ proj_j) {
  __shared__ float xl[128];
  const int bid = blockIdx.x;            // bid = b*512 + l
  const int l = bid & 511;
  const int h = threadIdx.x;
  const int idx = h >> 1;
  float freq = expf(-0.14391156832f * (float)idx);   // 10000^(-2*idx/128)
  float ang = (float)l * freq;
  float emb = (h & 1) ? cosf(ang) : sinf(ang);
  xl[h] = inputs[(size_t)bid * 128 + h] + emb;
  __syncthreads();
  float si = Whb[h], sj = 0.0f;
  const float* wr = Whw + h * 256;
  #pragma unroll 4
  for (int k = 0; k < 128; ++k) {
    si += xl[k] * wr[k];
    sj += xl[k] * wr[128 + k];
  }
  proj_i[(size_t)bid * 128 + h] = si;
  proj_j[(size_t)bid * 128 + h] = sj;
}

// ---- fused main: rel_proj GEMM -> +proj_i+proj_j -> tanh -> Wo GEMM -> mask/tril ----
// 512 threads = 16 waves; wave w owns position-row i = i0+w of the 16x16 (i,j) tile.
// LDS = 136KB -> two workgroups per WGP (8 waves/SIMD) for latency hiding.
__global__ __launch_bounds__(512)
void mhs_main_kernel(const float* __restrict__ rel,          // (512,512,256) fp32
                     const float* __restrict__ proj_i,       // (2,512,128) fp32 (bias folded)
                     const float* __restrict__ proj_j,       // (2,512,128) fp32
                     const unsigned short* __restrict__ whw_bf, // [128][256] bf16 bits
                     const unsigned short* __restrict__ wo_bf,  // [64][128] bf16 bits
                     const float* __restrict__ mask,         // (2,512)
                     const float* __restrict__ Wob,          // (49)
                     float* __restrict__ out) {              // (2,49,512,512)
  extern __shared__ char smem[];
  unsigned short* whw  = (unsigned short*)smem;             // 64KB [h=128][k=256] bf16 (LDS offset 0)
  unsigned short* hijb = (unsigned short*)(smem + 65536);   // 64KB [pos=256][h=128] bf16
  float* pj            = (float*)(smem + 131072);           // 8KB  [16][128] fp32

  const int tid   = threadIdx.x;
  const int wave  = tid >> 5;      // 0..15
  const int lane  = tid & 31;
  const int lhalf = lane >> 4;     // 0/1
  const int l16   = lane & 15;
  const int i0 = blockIdx.y * 16;
  const int j0 = blockIdx.x * 16;

  // ---- stage Wh_w (bf16, 64KB) into LDS via Tensor Data Mover when available ----
#if __has_builtin(__builtin_amdgcn_tensor_load_to_lds)
  if (wave == 0) {
    // D# group0: count=1, lds_addr=0 (whw at dynamic-LDS base), global_addr, type=2
    const unsigned long long ga = (unsigned long long)whw_bf;
    u32x4 g0;
    g0.x = 1u;                                               // count=1, is_restore=0
    g0.y = 0u;                                               // lds_addr (bytes)
    g0.z = (unsigned)(ga & 0xFFFFFFFFu);                     // global_addr[31:0]
    g0.w = ((unsigned)(ga >> 32) & 0x01FFFFFFu) | 0x80000000u; // global_addr[56:32] | type=2
    // D# group1: data_size=2B, 1 row of 32768 elements, tile == tensor
    i32x8 g1;
    g1[0] = 0x00010000;        // workgroup_mask=0, data_size=1 (2 bytes)
    g1[1] = (int)0x80000000;   // tensor_dim0 = 32768 (bits 79:48, low half here)
    g1[2] = 0x00010000;        // tensor_dim0 hi=0; tensor_dim1 = 1
    g1[3] = (int)0x80000000;   // tensor_dim1 hi=0; tile_dim0 = 32768
    g1[4] = 1;                 // tile_dim1 = 1, tile_dim2 = 0
    g1[5] = 32768;             // tensor_dim0_stride = 32768 (low 32)
    g1[6] = 0;                 // stride hi / dim1_stride
    g1[7] = 0;
    i32x4 gz4 = {0, 0, 0, 0};               // groups 2/3 unused (2-D tensor)
    i32x8 gz8 = {0, 0, 0, 0, 0, 0, 0, 0};   // 6-arg toolchain extra group
    __builtin_amdgcn_tensor_load_to_lds(g0, g1, gz4, gz4, gz8, 0);
  }
  __builtin_amdgcn_s_wait_tensorcnt(0);
#else
  {
    const uint4* src = (const uint4*)whw_bf;
    uint4* dst = (uint4*)whw;
    for (int t = tid; t < (128 * 256) / 8; t += 512) dst[t] = src[t];
  }
#endif
  __syncthreads();

  // ---------------- GEMM1: R[16 pos][128 h] per wave = relRow(16x256) * Whw^T ----------------
  // A-frag rows: lanes 0-15 / 16-31 both carry M=0..15 (l16), K split by lane-half (contiguous k = 16*half + s).
  // Software pipeline: next K-step's 4x float4 held in distinct regs while WMMAs run.
  const float* arow = rel + (((size_t)(i0 + wave)) * 512 + (j0 + l16)) * 256;
  const int kb = lhalf * 16;
  float4 fa, fb, fc, fd;
  {
    const float4* aq0 = (const float4*)(arow + kb);
    fa = aq0[0]; fb = aq0[1]; fc = aq0[2]; fd = aq0[3];
  }
  v8f acc[8] = {};
  #pragma unroll
  for (int ks = 0; ks < 8; ++ks) {
    // convert current K-step's A data
    union { v16bf v; v2bf p[8]; } afr;
    afr.p[0] = pk2(fa.x, fa.y); afr.p[1] = pk2(fa.z, fa.w);
    afr.p[2] = pk2(fb.x, fb.y); afr.p[3] = pk2(fb.z, fb.w);
    afr.p[4] = pk2(fc.x, fc.y); afr.p[5] = pk2(fc.z, fc.w);
    afr.p[6] = pk2(fd.x, fd.y); afr.p[7] = pk2(fd.z, fd.w);
    // issue next K-step's loads (overlap with B-frag loads + WMMA below)
    if (ks < 7) {
      const float4* aqn = (const float4*)(arow + (ks + 1) * 32 + kb);
      __builtin_prefetch(arow + (ks + 1) * 32 + kb + 32, 0, 1);
      fa = aqn[0]; fb = aqn[1]; fc = aqn[2]; fd = aqn[3];
    }
    // batch all 8 B-fragments, then run 8 back-to-back WMMAs
    v16bf bfr8[8];
    #pragma unroll
    for (int n = 0; n < 8; ++n)
      bfr8[n] = *(const v16bf*)(whw + (n * 16 + l16) * 256 + ks * 32 + kb);
    #pragma unroll
    for (int n = 0; n < 8; ++n)
      acc[n] = __builtin_amdgcn_wmma_f32_16x16x32_bf16(false, afr.v, false, bfr8[n],
                                                       (short)0, acc[n], false, false);
  }

  // ---------------- per-batch: broadcast add, tanh, GEMM2, epilogue ----------------
  const int i = i0 + wave;                  // this wave's output row
  for (int b = 0; b < 2; ++b) {
    __syncthreads();                        // pj reuse across b
    for (int t = tid; t < 16 * 128; t += 512)
      pj[t] = proj_j[(((size_t)b * 512) + j0 + (t >> 7)) * 128 + (t & 127)];
    // per-lane proj_i values for this wave's row (uniform row, 8 h-columns per lane)
    float piv[8];
    #pragma unroll
    for (int n = 0; n < 8; ++n)
      piv[n] = proj_i[(((size_t)b * 512) + i) * 128 + n * 16 + l16];
    __syncthreads();

    // tanh epilogue of GEMM1 -> bf16 hij rows in LDS (wave-local rows)
    #pragma unroll
    for (int n = 0; n < 8; ++n) {
      const int h = n * 16 + l16;
      #pragma unroll
      for (int v = 0; v < 8; ++v) {
        const int jl = v + 8 * lhalf;             // C-layout: M = v + 8*(lane/16)
        const int r = wave * 16 + jl;             // pos index (il = wave)
        float val = acc[n][v] + piv[n] + pj[jl * 128 + h];
        float t = fast_tanh(val);
        hijb[r * 128 + h] = (unsigned short)((__float_as_uint(t) + 0x8000u) >> 16);
      }
    }
    // hij rows of this wave are consumed only by this wave -> no barrier needed.

    // GEMM2: out(c x pos) = Wo(64x128) * hij^T ; N-tile = this wave's i-row (16 contiguous j)
    // B-fragments load directly as bf16 from LDS -- zero conversion VALU here.
    const float mcol = mask[b * 512 + j0 + l16];
    const float mrow = mask[b * 512 + i];
    v8f acc2[4] = {};
    #pragma unroll
    for (int k = 0; k < 4; ++k) {
      const int kk = k * 32 + kb;
      const int pos = wave * 16 + l16;
      v16bf bfrag = *(const v16bf*)(hijb + pos * 128 + kk);
      v16bf wf[4];
      #pragma unroll
      for (int mc = 0; mc < 4; ++mc)
        wf[mc] = *(const v16bf*)(wo_bf + (mc * 16 + l16) * 128 + kk);
      #pragma unroll
      for (int mc = 0; mc < 4; ++mc)
        acc2[mc] = __builtin_amdgcn_wmma_f32_16x16x32_bf16(false, wf[mc], false, bfrag,
                                                           (short)0, acc2[mc], false, false);
    }
    #pragma unroll
    for (int mc = 0; mc < 4; ++mc) {
      #pragma unroll
      for (int v = 0; v < 8; ++v) {
        const int c = mc * 16 + v + 8 * lhalf;
        if (c < 49) {
          const int j = j0 + l16;
          float val = acc2[mc][v] + Wob[c];
          val = val * mrow + (-NEGV) * (1.0f - mrow);
          val = val * mcol + (-NEGV) * (1.0f - mcol);
          if (i > j) val -= NEGV;                 // tril(k=-1) * NEG
          out[((((size_t)b * 49) + c) * 512 + i) * 512 + j] = val;
        }
      }
    }
  }
}

extern "C" void kernel_launch(void* const* d_in, const int* in_sizes, int n_in,
                              void* d_out, int out_size, void* d_ws, size_t ws_size,
                              hipStream_t stream) {
  const float* inputs = (const float*)d_in[0];
  const float* mask   = (const float*)d_in[1];
  const float* Whw    = (const float*)d_in[2];   // (128, 256)
  const float* Whb    = (const float*)d_in[3];   // (128)
  const float* Wow    = (const float*)d_in[4];   // (49, 128)
  const float* Wob    = (const float*)d_in[5];   // (49)
  const float* rel    = (const float*)d_in[6];   // (512, 512, 256)
  float* out = (float*)d_out;

  // workspace layout (~1.1 MB)
  float* proj_i = (float*)d_ws;                        // 2*512*128 fp32
  float* proj_j = proj_i + 2 * 512 * 128;              // 2*512*128 fp32
  unsigned short* whw_bf = (unsigned short*)(proj_j + 2 * 512 * 128); // 128*256 bf16
  unsigned short* wo_bf  = whw_bf + 128 * 256;         // 64*128 bf16

  prep_weights_kernel<<<dim3(1), dim3(256), 0, stream>>>(Whw, Wow, whw_bf, wo_bf);
  prep_proj_kernel<<<dim3(2 * 512), dim3(128), 0, stream>>>(inputs, Whw, Whb, proj_i, proj_j);

  const size_t shmem = 65536 + 65536 + 8192;  // Whw bf16 + hij bf16 + proj_j tile = 136KB
  mhs_main_kernel<<<dim3(32, 32), dim3(512), shmem, stream>>>(
      rel, proj_i, proj_j, whw_bf, wo_bf, mask, Wob, out);
}
